// SSL_6811818132018
// MI455X (gfx1250) — matrix operand order repaired and checked by reading, not compiled, not run on hardware
//
#include <hip/hip_runtime.h>
#include <hip/hip_bf16.h>

// ---------------- types for WMMA fragments ----------------
typedef __attribute__((ext_vector_type(16))) __bf16        v16bf;
typedef __attribute__((ext_vector_type(8)))  float         v8f;
typedef __attribute__((ext_vector_type(8)))  unsigned int  v8u;
typedef __attribute__((ext_vector_type(4)))  unsigned int  v4u;

#define EMBED_D 256
#define K_DIM   512   // 2*EMBED_D
#define HID     256
#define M_TILE  64    // pairs per workgroup

// gfx1250 async global->LDS path (ASYNCcnt-tracked), if this toolchain exposes it
#if __has_builtin(__builtin_amdgcn_global_load_async_to_lds_b128) && \
    __has_builtin(__builtin_amdgcn_s_wait_asynccnt)
#define SSL_USE_ASYNC_LDS 1
#define AS1 __attribute__((address_space(1)))
#define AS3 __attribute__((address_space(3)))
// builtin expects pointers to 16-byte int vectors (per hipcc diagnostic)
typedef int v4i_gcc __attribute__((__vector_size__(4 * sizeof(int))));
#else
#define SSL_USE_ASYNC_LDS 0
#endif

__device__ __forceinline__ unsigned short f32_to_bf16(float f) {
    unsigned int u = __builtin_bit_cast(unsigned int, f);
    // round-to-nearest-even
    unsigned int r = u + 0x7FFFu + ((u >> 16) & 1u);
    return (unsigned short)(r >> 16);
}

__device__ __forceinline__ unsigned int pack_bf16x2(float lo, float hi) {
    return (unsigned int)f32_to_bf16(lo) | ((unsigned int)f32_to_bf16(hi) << 16);
}

// ---------------- prep: W1 f32 [512][256] -> bf16 transposed [256][512]; zero out ----------------
__global__ void ssl_prep_kernel(const float* __restrict__ W1,
                                unsigned short* __restrict__ w1t,
                                float* __restrict__ out) {
    int idx = blockIdx.x * 256 + threadIdx.x;   // 512 blocks -> 131072 threads
    if (idx == 0) out[0] = 0.0f;
    if (idx < K_DIM * HID) {
        int k = idx >> 8;       // row in W1 (K)
        int n = idx & 255;      // col in W1 (N)
        w1t[n * K_DIM + k] = f32_to_bf16(W1[idx]);
    }
}

// ---------------- main fused kernel ----------------
__global__ __launch_bounds__(256)
void ssl_pairmlp_kernel(const float* __restrict__ embed,
                        const unsigned short* __restrict__ w1t,   // bf16 [256][512]
                        const float* __restrict__ bias1,
                        const float* __restrict__ W2,             // [256][2]
                        const float* __restrict__ bias2,
                        const int*   __restrict__ idx1,
                        const int*   __restrict__ idx2,
                        const int*   __restrict__ labels,
                        float* __restrict__ out,
                        int npairs, float invP) {
    // LDS layout:
    //   [0      .. 65536) : X tile, 64 rows x 256 uints (2 bf16 each) = 64x512 bf16
    //   [65536  .. 81920) : W1^T chunk, 256 rows x 16 uints = 256 x 32 bf16
    //   phase 3 reuses [0 .. 65792) as h1 f32 [64][257]
    __shared__ __align__(16) unsigned char smem[81920];
    __shared__ float bsum;

    unsigned int* xlds = (unsigned int*)smem;             // [64][256] uints
    unsigned int* wlds = (unsigned int*)(smem + 65536);   // [256][16] uints
    float*        h1   = (float*)smem;                    // [64][257] floats

    const int t    = threadIdx.x;
    const int lane = t & 31;
    const int wave = t >> 5;
    const int blockBase = blockIdx.x * M_TILE;

    // ---- Phase 1: gather two embeddings per pair, cvt bf16, stage X tile ----
    {
        int p   = t >> 2;   // 0..63 pair within tile
        int seg = t & 3;    // 4 segments of 128 source floats
        int pg  = blockBase + p;
        int safe = (pg < npairs) ? pg : 0;
        int i1 = idx1[safe];
        int i2 = idx2[safe];
        const float* src = (seg < 2)
            ? (embed + (size_t)i1 * EMBED_D + seg * 128)
            : (embed + (size_t)i2 * EMBED_D + (seg - 2) * 128);
        unsigned int* dst = xlds + p * 256 + seg * 64;    // 64 uints per segment
        #pragma unroll 8
        for (int j = 0; j < 32; ++j) {
            float4 v = ((const float4*)src)[j];
            dst[2 * j]     = pack_bf16x2(v.x, v.y);
            dst[2 * j + 1] = pack_bf16x2(v.z, v.w);
        }
    }

    // ---- Phase 2: X(64x512) @ W1(512x256) via v_wmma_f32_16x16x32_bf16 ----
    const int mb     = wave & 3;        // M-block 0..3 (16 rows each)
    const int nhalf  = wave >> 2;       // N half: 0 -> cols 0..127, 1 -> 128..255
    const int khalf  = lane >> 4;       // CDNA5 16-bit fragment lane K-half
    const int lane16 = lane & 15;

    const unsigned int* xrow = xlds + (mb * 16 + lane16) * 256;  // A: row = lane%16

    v8f acc[8] = {};

    for (int kc = 0; kc < 16; ++kc) {           // 16 chunks of K=32
        __syncthreads();                         // X ready / previous wlds reads done
        {   // stage W1^T chunk: thread t copies row n=t, 16 uints (32 bf16) contiguous
            const unsigned int* wsrc = (const unsigned int*)w1t + t * (K_DIM / 2) + kc * 16;
            unsigned int*       wdst = wlds + t * 16;
#if SSL_USE_ASYNC_LDS
            // VGPR-free ASYNCcnt-tracked copy: global_load_async_to_lds_b128 x4
            #pragma unroll
            for (int j = 0; j < 4; ++j)
                __builtin_amdgcn_global_load_async_to_lds_b128(
                    (AS1 v4i_gcc*)(wsrc + 4 * j), (AS3 v4i_gcc*)(wdst + 4 * j), 0, 0);
            __builtin_amdgcn_s_wait_asynccnt(0);
#else
            #pragma unroll
            for (int j = 0; j < 4; ++j)
                ((v4u*)wdst)[j] = ((const v4u*)wsrc)[j];
#endif
        }
        __syncthreads();

        // A fragment (16x32 bf16): V0-3 = K khalf*8+0..7, V4-7 = K 16+khalf*8+0..7
        const int kb2 = kc * 16;  // uint offset of chunk in X row
        v4u a0 = *(const v4u*)(xrow + kb2 + khalf * 4);
        v4u a1 = *(const v4u*)(xrow + kb2 + 8 + khalf * 4);
        v8u araw = {a0.x, a0.y, a0.z, a0.w, a1.x, a1.y, a1.z, a1.w};
        v16bf afrag = __builtin_bit_cast(v16bf, araw);

        #pragma unroll
        for (int i = 0; i < 8; ++i) {
            int n0 = (nhalf * 8 + i) * 16 + lane16;          // B: col = lane%16
            const unsigned int* wrow = wlds + n0 * 16 + khalf * 4;  // K khalf*8+0..7, 16+khalf*8+0..7
            v4u b0 = *(const v4u*)(wrow);
            v4u b1 = *(const v4u*)(wrow + 8);
            v8u braw = {b0.x, b0.y, b0.z, b0.w, b1.x, b1.y, b1.z, b1.w};
            v16bf bfrag = __builtin_bit_cast(v16bf, braw);
            acc[i] = __builtin_amdgcn_wmma_f32_16x16x32_bf16(
                false, afrag, false, bfrag, (short)0, acc[i], false, false);
        }
    }

    __syncthreads();   // all WMMA reads of X done before h1 overwrites it

    // ---- bias + ReLU, write h1 tile (stride 257 to avoid bank conflicts) ----
    #pragma unroll
    for (int i = 0; i < 8; ++i) {
        int n0 = (nhalf * 8 + i) * 16 + lane16;
        float bv = bias1[n0];
        #pragma unroll
        for (int v = 0; v < 8; ++v) {
            int r = mb * 16 + v + 8 * (lane >> 4);   // C/D layout: row = VGPR + 8*(lane/16)
            float h = acc[i][v] + bv;
            h1[r * 257 + n0] = h > 0.0f ? h : 0.0f;
        }
    }

    if (t == 0) bsum = 0.0f;
    __syncthreads();

    // ---- Phase 3: head GEMV (256->2), log-softmax, NLL, block reduce ----
    if (t < M_TILE) {
        int pg = blockBase + t;
        if (pg < npairs) {
            float l0 = bias2[0], l1 = bias2[1];
            const float* hr = h1 + t * 257;
            #pragma unroll 4
            for (int n = 0; n < HID; ++n) {
                float h = hr[n];
                l0 = fmaf(h, W2[2 * n], l0);
                l1 = fmaf(h, W2[2 * n + 1], l1);
            }
            float m   = fmaxf(l0, l1);
            float lse = m + logf(expf(l0 - m) + expf(l1 - m));
            int lab   = labels[pg];
            float nll = lse - (lab == 0 ? l0 : l1);
            atomicAdd(&bsum, nll);
        }
    }
    __syncthreads();
    if (t == 0) atomicAdd(out, bsum * invP);
}

extern "C" void kernel_launch(void* const* d_in, const int* in_sizes, int n_in,
                              void* d_out, int out_size, void* d_ws, size_t ws_size,
                              hipStream_t stream) {
    const float* embed = (const float*)d_in[0];
    const float* W1    = (const float*)d_in[1];
    const float* b1    = (const float*)d_in[2];
    const float* W2    = (const float*)d_in[3];
    const float* b2    = (const float*)d_in[4];
    const int*   i1    = (const int*)d_in[5];
    const int*   i2    = (const int*)d_in[6];
    const int*   lab   = (const int*)d_in[7];
    float*       out   = (float*)d_out;

    int P = in_sizes[5];
    unsigned short* w1t = (unsigned short*)d_ws;   // 512*256*2 = 256 KB bf16

    ssl_prep_kernel<<<512, 256, 0, stream>>>(W1, w1t, out);

    int nblocks = (P + M_TILE - 1) / M_TILE;
    ssl_pairmlp_kernel<<<nblocks, 256, 0, stream>>>(
        embed, w1t, b1, W2, b2, i1, i2, lab, out, P, 1.0f / (float)P);
}
